// NMT_33870112096567
// MI455X (gfx1250) — matrix operand order repaired and checked by reading, not compiled, not run on hardware
//
#include <hip/hip_runtime.h>
#include <hip/hip_bf16.h>
#include <math.h>

typedef __bf16 bf16_t;
typedef __attribute__((ext_vector_type(16))) __bf16 v16bf;
typedef __attribute__((ext_vector_type(8)))  __bf16 v8bf;
typedef __attribute__((ext_vector_type(8)))  float  v8f;

#define B_      32
#define S_      512
#define T_      256
#define E_      128
#define H_      256
#define VOCAB_  32000

__device__ __forceinline__ bf16_t f2bf(float f) {
  union { float f; unsigned u; } v; v.f = f;
  unsigned r = v.u + 0x7FFFu + ((v.u >> 16) & 1u);   // round-to-nearest-even
  union { unsigned short u; bf16_t b; } o;
  o.u = (unsigned short)(r >> 16);
  return o.b;
}

__device__ __forceinline__ float sigmoidf_(float x) { return 1.0f / (1.0f + __expf(-x)); }

// Opaque zero (SGPR): added to weight-fragment element indices inside the
// recurrence loops. Being loop-variant from the compiler's viewpoint, it
// blocks LICM/CSE of the loop-invariant weight loads (which previously blew
// up registers and spilled to scratch) -- while keeping the base pointer's
// provenance intact so loads stay global_load (SADDR form), not flat_load.
__device__ __forceinline__ int opaque_zero() {
  int x = 0;
  asm volatile("" : "+s"(x));
  return x;
}

// ---------------------------------------------------------------------------
// WMMA fragment helpers (bf16 A/B, f32 C/D), CDNA5 16x16x32 layouts.
// A (16xK row-major): lane<16: elems[0..7]=K(k0..k0+7), elems[8..15]=K(k0+16..k0+23)
//                     lane>=16: +8 on both K windows.
__device__ __forceinline__ v16bf load_a(const bf16_t* A, int ldk, int k0, int lane) {
  int m  = lane & 15;
  int kb = (lane >> 4) << 3;                 // 0 or 8
  const bf16_t* base = A + m * ldk + k0 + kb;
  v8bf lo = *(const v8bf*)(base);
  v8bf hi = *(const v8bf*)(base + 16);
  return __builtin_shufflevector(lo, hi, 0,1,2,3,4,5,6,7,8,9,10,11,12,13,14,15);
}

// B (KxN) sourced from row-major weight W (N_total x K): B[k][n] = W[n][k].
// lane<16: N=n0+lane, elems[0..7]=K(k0..k0+7), elems[8..15]=K(k0+16..k0+23); lane>=16: N=n0+lane-16, +8 on both K windows.
// z is 0 (opaque) inside recurrence loops to defeat hoisting.
__device__ __forceinline__ v16bf load_b(const bf16_t* W, int n0, int k0, int ldk, int lane, int z = 0) {
  int n  = n0 + (lane & 15);
  int kb = k0 + ((lane >> 4) << 3);          // 0 or +8
  const bf16_t* base = W + ((size_t)n * ldk + kb + z);
  v8bf lo = *(const v8bf*)(base);
  v8bf hi = *(const v8bf*)(base + 16);
  return __builtin_shufflevector(lo, hi, 0,1,2,3,4,5,6,7,8,9,10,11,12,13,14,15);
}

__device__ __forceinline__ v8f wmma_bf(v16bf a, v16bf b, v8f c) {
  return __builtin_amdgcn_wmma_f32_16x16x32_bf16(false, a, false, b, (short)0, c, false, false);
}

// ---------------------------------------------------------------------------
__global__ void cvt_bf16_kernel(const float* __restrict__ src, bf16_t* __restrict__ dst, int n) {
  int i = blockIdx.x * blockDim.x + threadIdx.x;
  if (i < n) dst[i] = f2bf(src[i]);
}

// ---------------------------------------------------------------------------
// Bidirectional LSTM encoder. grid = 4: blockIdx.x = dir + 2*btile.
// 512 threads = 16 waves; wave w owns hidden columns [16w, 16w+16).
// Per step: gates = bias + x_t@Wih^T + h@Whh^T via WMMA; cell update in regs.
__global__ void lstm_kernel(const int* __restrict__ enc_inputs, const bf16_t* __restrict__ emb_bf,
                            const bf16_t* __restrict__ Wih_f, const bf16_t* __restrict__ Whh_f,
                            const float* __restrict__ bih_f, const float* __restrict__ bhh_f,
                            const bf16_t* __restrict__ Wih_b, const bf16_t* __restrict__ Whh_b,
                            const float* __restrict__ bih_b, const float* __restrict__ bhh_b,
                            float* __restrict__ enc_out, bf16_t* __restrict__ enc_out_bf) {
  __shared__ __align__(16) bf16_t xs[16 * E_];   // x_t tile (bf16)
  __shared__ __align__(16) bf16_t hsm[16 * H_];  // h_{t-1} tile (bf16)

  int dir = blockIdx.x & 1;
  int bt  = blockIdx.x >> 1;
  const bf16_t* Wih = dir ? Wih_b : Wih_f;
  const bf16_t* Whh = dir ? Whh_b : Whh_f;
  const float*  bih = dir ? bih_b : bih_f;
  const float*  bhh = dir ? bhh_b : bhh_f;

  int w     = threadIdx.x >> 5;
  int lane  = threadIdx.x & 31;
  int mbase = (lane >> 4) << 3;
  int ncol  = lane & 15;

  // biases are tiny: keep in registers across the whole loop
  float bsum[4];
  #pragma unroll
  for (int gi = 0; gi < 4; gi++) {
    int n = gi * H_ + w * 16 + ncol;
    bsum[gi] = bih[n] + bhh[n];
  }

  for (int i = threadIdx.x; i < 16 * H_; i += 512) hsm[i] = f2bf(0.0f);
  v8f cstate = {};
  __syncthreads();

  for (int t = 0; t < S_; t++) {
    int tw = dir ? (S_ - 1 - t) : t;
    int z  = opaque_zero();   // blocks weight-load hoisting across time steps

    // stage x_t: gather embedding rows (bf16)
    {
      int row = threadIdx.x >> 5;
      int c4  = (threadIdx.x & 31) * 4;
      int tok = enc_inputs[(bt * 16 + row) * S_ + tw];
      const bf16_t* src = emb_bf + (size_t)tok * E_ + c4;
      #pragma unroll
      for (int i = 0; i < 4; i++) xs[row * E_ + c4 + i] = src[i];
    }
    __syncthreads();

    // 4 gate tiles: i,f,g,o for hidden cols [16w,16w+16)
    v8f g[4];
    #pragma unroll
    for (int gi = 0; gi < 4; gi++) {
      int n0 = gi * H_ + w * 16;
      float bias = bsum[gi];
      v8f acc = { bias, bias, bias, bias, bias, bias, bias, bias };
      #pragma unroll
      for (int kk = 0; kk < E_; kk += 32)
        acc = wmma_bf(load_a(xs, E_, kk, lane), load_b(Wih, n0, kk, E_, lane, z), acc);
      #pragma unroll
      for (int kk = 0; kk < H_; kk += 32)
        acc = wmma_bf(load_a(hsm, H_, kk, lane), load_b(Whh, n0, kk, H_, lane, z), acc);
      g[gi] = acc;
    }
    __syncthreads();  // all reads of hsm done before rewrite

    int col = w * 16 + ncol;
    #pragma unroll
    for (int r = 0; r < 8; r++) {
      float ig = sigmoidf_(g[0][r]);
      float fg = sigmoidf_(g[1][r]);
      float gg = tanhf(g[2][r]);
      float og = sigmoidf_(g[3][r]);
      float cv = fg * cstate[r] + ig * gg;
      cstate[r] = cv;
      float hv = og * tanhf(cv);
      int m = mbase + r;
      hsm[m * H_ + col] = f2bf(hv);
      size_t oidx = ((size_t)(bt * 16 + m) * S_ + tw) * (2 * H_) + dir * H_ + col;
      enc_out[oidx]    = hv;
      enc_out_bf[oidx] = f2bf(hv);
    }
    __syncthreads();
  }
}

// ---------------------------------------------------------------------------
// Uenc[b,s,h] = enc_out[b,s,:] . attn_U[h,:]   -> (B*S, H) GEMM, K=512
// grid = (B*S)/16 = 1024 blocks, 256 threads = 8 waves, 2 n-tiles per wave.
__global__ void uenc_kernel(const bf16_t* __restrict__ enc_out_bf,
                            const bf16_t* __restrict__ attnU_bf,
                            float* __restrict__ Uenc) {
  int mt   = blockIdx.x;
  int w    = threadIdx.x >> 5;
  int lane = threadIdx.x & 31;
  int mbase = (lane >> 4) << 3;
  int ncol  = lane & 15;
  const bf16_t* A = enc_out_bf + (size_t)mt * 16 * (2 * H_);

  for (int nt = w; nt < 16; nt += 8) {
    int n0 = nt * 16;
    v8f acc = {};
    #pragma unroll
    for (int kk = 0; kk < 2 * H_; kk += 32)
      acc = wmma_bf(load_a(A, 2 * H_, kk, lane), load_b(attnU_bf, n0, kk, 2 * H_, lane), acc);
    #pragma unroll
    for (int r = 0; r < 8; r++)
      Uenc[(size_t)(mt * 16 + mbase + r) * H_ + n0 + ncol] = acc[r];
  }
}

// ---------------------------------------------------------------------------
// Attention decoder. grid = 2 (batch tiles of 16). 512 threads = 16 waves.
__global__ void decoder_kernel(const int* __restrict__ dec_inputs, const bf16_t* __restrict__ dec_emb_bf,
                               const float* __restrict__ enc_out, const float* __restrict__ Uenc,
                               const bf16_t* __restrict__ attnW_bf,
                               const float* __restrict__ attn_v_w, const float* __restrict__ attn_v_b,
                               const float* __restrict__ attn_bias,
                               const bf16_t* __restrict__ Wz_bf, const bf16_t* __restrict__ Uz_bf, const bf16_t* __restrict__ Cz_bf,
                               const bf16_t* __restrict__ Wr_bf, const bf16_t* __restrict__ Ur_bf, const bf16_t* __restrict__ Cr_bf,
                               const bf16_t* __restrict__ Wd_bf, const bf16_t* __restrict__ Ud_bf, const bf16_t* __restrict__ Cd_bf,
                               float* __restrict__ out) {
  extern __shared__ char dsm[];
  float*  sc   = (float*)(dsm);                  // [16][512] scores -> attn
  float*  spre = (float*)(dsm + 32768);          // [16][256] s @ attn_W^T
  bf16_t* ctxb = (bf16_t*)(dsm + 49152);         // [16][512] context (bf16)
  bf16_t* sbf  = (bf16_t*)(dsm + 65536);         // [16][256] s (bf16)
  bf16_t* rsb  = (bf16_t*)(dsm + 73728);         // [16][256] r*s (bf16)
  bf16_t* ybf  = (bf16_t*)(dsm + 81920);         // [16][128] y_t (bf16)
  float*  stg  = (float*)(dsm + 86016);          // [512] attn_bias(256) | v_w(256)

  int bt    = blockIdx.x;
  int w     = threadIdx.x >> 5;
  int lane  = threadIdx.x & 31;
  int mbase = (lane >> 4) << 3;
  int ncol  = lane & 15;
  int n0    = w * 16;

  if (threadIdx.x < H_) {
    stg[threadIdx.x]      = attn_bias[threadIdx.x];
    stg[H_ + threadIdx.x] = attn_v_w[threadIdx.x];
  }
  float vb = attn_v_b[0];

  // s0 = enc_out[:, S-1, :H]  (wave w keeps its 16 columns in registers)
  v8f sfrag;
  {
    int col = n0 + ncol;
    #pragma unroll
    for (int r = 0; r < 8; r++) {
      int m = mbase + r;
      float sv = enc_out[((size_t)(bt * 16 + m) * S_ + (S_ - 1)) * (2 * H_) + col];
      sfrag[r] = sv;
      sbf[m * H_ + col] = f2bf(sv);
    }
  }
  __syncthreads();

  for (int t = 0; t < T_; t++) {
    int z = opaque_zero();   // blocks weight-load hoisting across time steps

    // stage y_t
    {
      int row = threadIdx.x >> 5;
      int c4  = (threadIdx.x & 31) * 4;
      int tok = dec_inputs[(bt * 16 + row) * T_ + t];
      const bf16_t* src = dec_emb_bf + (size_t)tok * E_ + c4;
      #pragma unroll
      for (int i = 0; i < 4; i++) ybf[row * E_ + c4 + i] = src[i];
    }
    __syncthreads();

    // spre = s @ attn_W^T
    {
      v8f acc = {};
      #pragma unroll
      for (int kk = 0; kk < H_; kk += 32)
        acc = wmma_bf(load_a(sbf, H_, kk, lane), load_b(attnW_bf, n0, kk, H_, lane, z), acc);
      #pragma unroll
      for (int r = 0; r < 8; r++) spre[(mbase + r) * H_ + n0 + ncol] = acc[r];
    }
    __syncthreads();

    // scores[b][sidx] = v . tanh(spre[b] + Uenc[b,sidx] + bias) + vb
    {
      int b   = threadIdx.x >> 5;
      int s0i = threadIdx.x & 31;
      for (int p = 0; p < 16; p++) {
        int sidx = s0i + 32 * p;
        const float* ue = Uenc + ((size_t)(bt * 16 + b) * S_ + sidx) * H_;
        float acc = 0.f;
        #pragma unroll 8
        for (int h = 0; h < H_; h++) {
          float val = spre[b * H_ + h] + ue[h] + stg[h];
          acc += tanhf(val) * stg[H_ + h];
        }
        sc[b * S_ + sidx] = acc + vb;
      }
    }
    __syncthreads();

    // softmax over sequence, wave w handles batch row w
    {
      float vals[16];
      float mx = -1e30f;
      #pragma unroll
      for (int q = 0; q < 16; q++) { vals[q] = sc[w * S_ + lane + 32 * q]; mx = fmaxf(mx, vals[q]); }
      #pragma unroll
      for (int o = 16; o > 0; o >>= 1) mx = fmaxf(mx, __shfl_xor(mx, o, 32));
      float sum = 0.f;
      #pragma unroll
      for (int q = 0; q < 16; q++) { vals[q] = __expf(vals[q] - mx); sum += vals[q]; }
      #pragma unroll
      for (int o = 16; o > 0; o >>= 1) sum += __shfl_xor(sum, o, 32);
      float inv = 1.0f / sum;
      #pragma unroll
      for (int q = 0; q < 16; q++) sc[w * S_ + lane + 32 * q] = vals[q] * inv;
    }
    __syncthreads();

    // ctx[b][k] = sum_s attn[b][s] * enc_out[b][s][k]
    {
      int b   = threadIdx.x >> 5;
      int k0i = threadIdx.x & 31;
      for (int p = 0; p < 16; p++) {
        int k = k0i + 32 * p;
        const float* eo = enc_out + ((size_t)(bt * 16 + b) * S_) * (2 * H_) + k;
        float acc = 0.f;
        #pragma unroll 8
        for (int s = 0; s < S_; s++) acc += sc[b * S_ + s] * eo[(size_t)s * (2 * H_)];
        ctxb[b * (2 * H_) + k] = f2bf(acc);
      }
    }
    __syncthreads();

    // z and r gates: sigmoid(y@W^T + s@U^T + ctx@C^T)
    v8f zf, rf;
    {
      v8f acc = {};
      #pragma unroll
      for (int kk = 0; kk < E_; kk += 32)     acc = wmma_bf(load_a(ybf,  E_,     kk, lane), load_b(Wz_bf, n0, kk, E_,     lane, z), acc);
      #pragma unroll
      for (int kk = 0; kk < H_; kk += 32)     acc = wmma_bf(load_a(sbf,  H_,     kk, lane), load_b(Uz_bf, n0, kk, H_,     lane, z), acc);
      #pragma unroll
      for (int kk = 0; kk < 2 * H_; kk += 32) acc = wmma_bf(load_a(ctxb, 2 * H_, kk, lane), load_b(Cz_bf, n0, kk, 2 * H_, lane, z), acc);
      #pragma unroll
      for (int r = 0; r < 8; r++) zf[r] = sigmoidf_(acc[r]);
    }
    {
      v8f acc = {};
      #pragma unroll
      for (int kk = 0; kk < E_; kk += 32)     acc = wmma_bf(load_a(ybf,  E_,     kk, lane), load_b(Wr_bf, n0, kk, E_,     lane, z), acc);
      #pragma unroll
      for (int kk = 0; kk < H_; kk += 32)     acc = wmma_bf(load_a(sbf,  H_,     kk, lane), load_b(Ur_bf, n0, kk, H_,     lane, z), acc);
      #pragma unroll
      for (int kk = 0; kk < 2 * H_; kk += 32) acc = wmma_bf(load_a(ctxb, 2 * H_, kk, lane), load_b(Cr_bf, n0, kk, 2 * H_, lane, z), acc);
      #pragma unroll
      for (int r = 0; r < 8; r++) rf[r] = sigmoidf_(acc[r]);
    }
    // rs = r * s
    {
      int col = n0 + ncol;
      #pragma unroll
      for (int r = 0; r < 8; r++) rsb[(mbase + r) * H_ + col] = f2bf(rf[r] * sfrag[r]);
    }
    __syncthreads();

    // s_tilde = tanh(y@Wd^T + (r*s)@Ud^T + ctx@Cd^T); s = (1-z)*s + z*s_tilde
    {
      v8f acc = {};
      #pragma unroll
      for (int kk = 0; kk < E_; kk += 32)     acc = wmma_bf(load_a(ybf,  E_,     kk, lane), load_b(Wd_bf, n0, kk, E_,     lane, z), acc);
      #pragma unroll
      for (int kk = 0; kk < H_; kk += 32)     acc = wmma_bf(load_a(rsb,  H_,     kk, lane), load_b(Ud_bf, n0, kk, H_,     lane, z), acc);
      #pragma unroll
      for (int kk = 0; kk < 2 * H_; kk += 32) acc = wmma_bf(load_a(ctxb, 2 * H_, kk, lane), load_b(Cd_bf, n0, kk, 2 * H_, lane, z), acc);
      int col = n0 + ncol;
      #pragma unroll
      for (int r = 0; r < 8; r++) {
        float st = tanhf(acc[r]);
        float sn = (1.0f - zf[r]) * sfrag[r] + zf[r] * st;
        sfrag[r] = sn;
        int m = mbase + r;
        sbf[m * H_ + col] = f2bf(sn);
        out[((size_t)(bt * 16 + m) * T_ + t) * H_ + col] = sn;
      }
    }
    __syncthreads();
  }
}

// ---------------------------------------------------------------------------
extern "C" void kernel_launch(void* const* d_in, const int* in_sizes, int n_in,
                              void* d_out, int out_size, void* d_ws, size_t ws_size,
                              hipStream_t stream) {
  (void)in_sizes; (void)n_in; (void)out_size; (void)ws_size;

  const int*   enc_inputs = (const int*)d_in[0];
  const int*   dec_inputs = (const int*)d_in[1];
  const float* enc_emb  = (const float*)d_in[2];
  const float* Wih_f    = (const float*)d_in[3];
  const float* Whh_f    = (const float*)d_in[4];
  const float* bih_f    = (const float*)d_in[5];
  const float* bhh_f    = (const float*)d_in[6];
  const float* Wih_b    = (const float*)d_in[7];
  const float* Whh_b    = (const float*)d_in[8];
  const float* bih_b    = (const float*)d_in[9];
  const float* bhh_b    = (const float*)d_in[10];
  const float* dec_emb  = (const float*)d_in[11];
  const float* attn_W   = (const float*)d_in[12];
  const float* attn_U   = (const float*)d_in[13];
  const float* attn_v_w = (const float*)d_in[14];
  const float* attn_v_b = (const float*)d_in[15];
  const float* attn_bias= (const float*)d_in[16];
  const float* Wz = (const float*)d_in[17];
  const float* Uz = (const float*)d_in[18];
  const float* Cz = (const float*)d_in[19];
  const float* Wr = (const float*)d_in[20];
  const float* Ur = (const float*)d_in[21];
  const float* Cr = (const float*)d_in[22];
  const float* Wd = (const float*)d_in[23];
  const float* Ud = (const float*)d_in[24];
  const float* Cd = (const float*)d_in[25];

  char* ws = (char*)d_ws;
  size_t off = 0;
  auto wsalloc = [&](size_t bytes) -> char* {
    char* p = ws + off;
    off += (bytes + 255) & ~(size_t)255;
    return p;
  };

  float*  enc_out    = (float*)wsalloc((size_t)B_ * S_ * 2 * H_ * 4);
  bf16_t* enc_out_bf = (bf16_t*)wsalloc((size_t)B_ * S_ * 2 * H_ * 2);
  float*  Uenc       = (float*)wsalloc((size_t)B_ * S_ * H_ * 4);
  bf16_t* enc_emb_bf = (bf16_t*)wsalloc((size_t)VOCAB_ * E_ * 2);
  bf16_t* dec_emb_bf = (bf16_t*)wsalloc((size_t)VOCAB_ * E_ * 2);
  bf16_t* Wih_f_bf   = (bf16_t*)wsalloc((size_t)4 * H_ * E_ * 2);
  bf16_t* Whh_f_bf   = (bf16_t*)wsalloc((size_t)4 * H_ * H_ * 2);
  bf16_t* Wih_b_bf   = (bf16_t*)wsalloc((size_t)4 * H_ * E_ * 2);
  bf16_t* Whh_b_bf   = (bf16_t*)wsalloc((size_t)4 * H_ * H_ * 2);
  bf16_t* attnW_bf   = (bf16_t*)wsalloc((size_t)H_ * H_ * 2);
  bf16_t* attnU_bf   = (bf16_t*)wsalloc((size_t)H_ * 2 * H_ * 2);
  bf16_t* Uz_bf      = (bf16_t*)wsalloc((size_t)H_ * H_ * 2);
  bf16_t* Ur_bf      = (bf16_t*)wsalloc((size_t)H_ * H_ * 2);
  bf16_t* Ud_bf      = (bf16_t*)wsalloc((size_t)H_ * H_ * 2);
  bf16_t* Cz_bf      = (bf16_t*)wsalloc((size_t)H_ * 2 * H_ * 2);
  bf16_t* Cr_bf      = (bf16_t*)wsalloc((size_t)H_ * 2 * H_ * 2);
  bf16_t* Cd_bf      = (bf16_t*)wsalloc((size_t)H_ * 2 * H_ * 2);
  bf16_t* Wz_bf      = (bf16_t*)wsalloc((size_t)H_ * E_ * 2);
  bf16_t* Wr_bf      = (bf16_t*)wsalloc((size_t)H_ * E_ * 2);
  bf16_t* Wd_bf      = (bf16_t*)wsalloc((size_t)H_ * E_ * 2);

  auto cvt = [&](const float* s, bf16_t* d, int n) {
    cvt_bf16_kernel<<<dim3((n + 511) / 512), dim3(512), 0, stream>>>(s, d, n);
  };
  cvt(enc_emb, enc_emb_bf, VOCAB_ * E_);
  cvt(dec_emb, dec_emb_bf, VOCAB_ * E_);
  cvt(Wih_f, Wih_f_bf, 4 * H_ * E_);
  cvt(Whh_f, Whh_f_bf, 4 * H_ * H_);
  cvt(Wih_b, Wih_b_bf, 4 * H_ * E_);
  cvt(Whh_b, Whh_b_bf, 4 * H_ * H_);
  cvt(attn_W, attnW_bf, H_ * H_);
  cvt(attn_U, attnU_bf, H_ * 2 * H_);
  cvt(Uz, Uz_bf, H_ * H_);  cvt(Ur, Ur_bf, H_ * H_);  cvt(Ud, Ud_bf, H_ * H_);
  cvt(Cz, Cz_bf, H_ * 2 * H_); cvt(Cr, Cr_bf, H_ * 2 * H_); cvt(Cd, Cd_bf, H_ * 2 * H_);
  cvt(Wz, Wz_bf, H_ * E_);  cvt(Wr, Wr_bf, H_ * E_);  cvt(Wd, Wd_bf, H_ * E_);

  // Bidirectional LSTM encoder: 4 persistent workgroups (dir x batch-tile)
  lstm_kernel<<<dim3(4), dim3(512), 0, stream>>>(
      enc_inputs, enc_emb_bf,
      Wih_f_bf, Whh_f_bf, bih_f, bhh_f,
      Wih_b_bf, Whh_b_bf, bih_b, bhh_b,
      enc_out, enc_out_bf);

  // Uenc = enc_out @ attn_U^T
  uenc_kernel<<<dim3((B_ * S_) / 16), dim3(256), 0, stream>>>(enc_out_bf, attnU_bf, Uenc);

  // Attention decoder: 2 persistent workgroups (batch tiles)
  size_t dec_lds = 88064;
  hipFuncSetAttribute((const void*)decoder_kernel,
                      hipFuncAttributeMaxDynamicSharedMemorySize, (int)dec_lds);
  decoder_kernel<<<dim3(2), dim3(512), dec_lds, stream>>>(
      dec_inputs, dec_emb_bf, enc_out, Uenc,
      attnW_bf, attn_v_w, attn_v_b, attn_bias,
      Wz_bf, Uz_bf, Cz_bf,
      Wr_bf, Ur_bf, Cr_bf,
      Wd_bf, Ud_bf, Cd_bf,
      (float*)d_out);
}
